// DbrxExperts_4020089389381
// MI455X (gfx1250) — matrix operand order repaired and checked by reading, not compiled, not run on hardware
//
#include <hip/hip_runtime.h>
#include <math.h>

#define T_TOK 4096
#define DDIM  2048
#define NEXP  8
#define IDIM  4096

#define BM 128          // block tile: tokens (padded slots)
#define BN 128          // block tile: output features
#define BK 32           // K step (WMMA K)
#define PITCH 40        // LDS row pitch in bf16 elements (32 data + 8 pad)
#define TILE_ELEMS (BM * PITCH)

typedef __attribute__((ext_vector_type(16))) __bf16 v16bf;
typedef __attribute__((ext_vector_type(8)))  __bf16 v8bf;
typedef __attribute__((ext_vector_type(4)))  __bf16 v4bf;
typedef __attribute__((ext_vector_type(8)))  float  v8f;
typedef __attribute__((ext_vector_type(4)))  float  v4f;

static __device__ inline int imin(int a, int b) { return a < b ? a : b; }

// ---------------------------------------------------------------------------
// WMMA bf16 16x16x32 fragment loads from LDS (wave32 layouts, ISA 7.12.2)
// A 16x32: lane<16 -> row m0+lane, K {0..7,16..23}; lane>=16 -> K {8..15,24..31}
// B 32x16: lane<16 -> col n0+lane, K {0..7,16..23}; lane>=16 -> K {8..15,24..31}
// ---------------------------------------------------------------------------
static __device__ inline v16bf frag_a_lds(const __bf16* s, int m0, int lane) {
  const int lr = lane & 15, hi = lane >> 4;
  const __bf16* p = s + (m0 + lr) * PITCH + hi * 8;
  v8bf lo = *(const v8bf*)p;
  v8bf hh = *(const v8bf*)(p + 16);
  v16bf a;
#pragma unroll
  for (int i = 0; i < 8; ++i) { a[i] = lo[i]; a[i + 8] = hh[i]; }
  return a;
}

static __device__ inline v16bf frag_b_lds(const __bf16* s, int n0, int lane) {
  const int lr = lane & 15, hi = lane >> 4;
  const __bf16* p = s + (n0 + lr) * PITCH + hi * 8;
  v8bf lo = *(const v8bf*)p;
  v8bf hh = *(const v8bf*)(p + 16);
  v16bf b;
#pragma unroll
  for (int i = 0; i < 8; ++i) { b[i] = lo[i]; b[i + 8] = hh[i]; }
  return b;
}

// ---------------------------------------------------------------------------
// Register-prefetched cooperative staging (128x32 tile per 256 threads).
// Load (global->regs) is issued before the compute phase; convert+store to the
// *other* LDS buffer afterwards, so load latency overlaps the WMMA stream.
// ---------------------------------------------------------------------------
struct F32Tile { v4f v[4]; };
struct Bf16Tile { v8bf v[2]; };

static __device__ inline void load_f32_tile(F32Tile& t, const float* __restrict__ src,
                                            size_t ld, int tid) {
#pragma unroll
  for (int i = 0; i < 4; ++i) {
    int idx = tid + i * 256;             // 0..1023 float4 chunks
    int row = idx >> 3;
    int kq  = idx & 7;
    t.v[i] = *(const v4f*)(src + (size_t)row * ld + kq * 4);
  }
}

static __device__ inline void load_f32_tile_gather(F32Tile& t, const float* __restrict__ X,
                                                   const int* toks, int k0, int tid) {
#pragma unroll
  for (int i = 0; i < 4; ++i) {
    int idx = tid + i * 256;
    int row = idx >> 3;
    int kq  = idx & 7;
    int tok = toks[row];
    t.v[i] = *(const v4f*)(X + (size_t)tok * DDIM + k0 + kq * 4);
  }
}

static __device__ inline void store_tile_bf16(__bf16* dst, const F32Tile& t, int tid) {
#pragma unroll
  for (int i = 0; i < 4; ++i) {
    int idx = tid + i * 256;
    int row = idx >> 3;
    int kq  = idx & 7;
    v4bf o;
#pragma unroll
    for (int c = 0; c < 4; ++c) o[c] = (__bf16)t.v[i][c];
    *(v4bf*)(dst + row * PITCH + kq * 4) = o;
  }
}

static __device__ inline void load_bf16_tile(Bf16Tile& t, const __bf16* __restrict__ src,
                                             size_t ld, int tid) {
#pragma unroll
  for (int i = 0; i < 2; ++i) {
    int idx = tid + i * 256;             // 0..511 8-elem chunks
    int row = idx >> 2;
    int ks  = idx & 3;
    t.v[i] = *(const v8bf*)(src + (size_t)row * ld + ks * 8);
  }
}

static __device__ inline void store_tile_copy(__bf16* dst, const Bf16Tile& t, int tid) {
#pragma unroll
  for (int i = 0; i < 2; ++i) {
    int idx = tid + i * 256;
    int row = idx >> 2;
    int ks  = idx & 3;
    *(v8bf*)(dst + row * PITCH + ks * 8) = t.v[i];
  }
}

// ---------------------------------------------------------------------------
// Zero output + expert counters (output accumulates across expert passes).
// ---------------------------------------------------------------------------
__global__ __launch_bounds__(256) void moe_zero_kernel(float* __restrict__ out,
                                                       int* __restrict__ counts) {
  size_t i = (size_t)blockIdx.x * 256 + threadIdx.x;   // grid covers T*D/4 exactly
  v4f z = {};
  ((v4f*)out)[i] = z;
  if (i < NEXP) counts[i] = 0;
}

// ---------------------------------------------------------------------------
// Router: exact fp32 logits, pairwise softmax of top-2 (== softmax+renorm),
// dense gates[T][8], atomic append of token ids to per-expert lists.
// ---------------------------------------------------------------------------
__global__ __launch_bounds__(256) void moe_router_kernel(
    const float* __restrict__ X, const float* __restrict__ WR,
    float* __restrict__ gates, int* __restrict__ counts, int* __restrict__ list) {
  const int tid  = threadIdx.x;
  const int lane = tid & 31;
  const int wave = tid >> 5;
  const int t = blockIdx.x * 8 + wave;

  float part[NEXP];
#pragma unroll
  for (int e = 0; e < NEXP; ++e) part[e] = 0.0f;

  for (int d = lane; d < DDIM; d += 32) {
    float xv = X[(size_t)t * DDIM + d];
#pragma unroll
    for (int e = 0; e < NEXP; ++e) part[e] += xv * WR[(size_t)e * DDIM + d];
  }
#pragma unroll
  for (int e = 0; e < NEXP; ++e)
    for (int off = 16; off > 0; off >>= 1)
      part[e] += __shfl_xor(part[e], off, 32);

  if (lane == 0) {
    int i1 = 0;
#pragma unroll
    for (int e = 1; e < NEXP; ++e) if (part[e] > part[i1]) i1 = e;
    int i2 = (i1 == 0) ? 1 : 0;
#pragma unroll
    for (int e = 0; e < NEXP; ++e)
      if (e != i1 && part[e] > part[i2]) i2 = e;
    float ex = expf(part[i2] - part[i1]);   // l2 <= l1: no overflow
    float g1 = 1.0f / (1.0f + ex);
    float g2 = ex * g1;
#pragma unroll
    for (int e = 0; e < NEXP; ++e) gates[(size_t)t * NEXP + e] = 0.0f;
    gates[(size_t)t * NEXP + i1] = g1;
    gates[(size_t)t * NEXP + i2] = g2;
    int p1 = atomicAdd(&counts[i1], 1);
    list[(size_t)i1 * T_TOK + p1] = t;
    int p2 = atomicAdd(&counts[i2], 1);
    list[(size_t)i2 * T_TOK + p2] = t;
  }
}

// ---------------------------------------------------------------------------
// Pass 1 (per expert, gathered): Hc[slot,:] = silu(x_tok @ W1^T) * (x_tok @ V1^T)
// Double-buffered LDS pipeline; wave tile 64(M)x32(N), dual accumulators.
// ---------------------------------------------------------------------------
__global__ __launch_bounds__(256) void moe_h_kernel(
    const float* __restrict__ X, const float* __restrict__ W1,
    const float* __restrict__ V1, const int* __restrict__ cntp,
    const int* __restrict__ list_e, __bf16* __restrict__ Hc) {
  __shared__ __attribute__((aligned(16))) __bf16 sA[2][TILE_ELEMS];
  __shared__ __attribute__((aligned(16))) __bf16 sB1[2][TILE_ELEMS];
  __shared__ __attribute__((aligned(16))) __bf16 sB2[2][TILE_ELEMS];
  __shared__ int toks[BM];

  const int tid = threadIdx.x, lane = tid & 31, wave = tid >> 5;
  const int waveM = wave & 1, waveN = wave >> 1;       // 2 x 4 wave grid
  const int cnt = *cntp;
  const int m_block = blockIdx.y * BM;
  if (m_block >= cnt) return;                          // uniform early exit
  const int n_block = blockIdx.x * BN;

  if (tid < BM) toks[tid] = list_e[imin(m_block + tid, cnt - 1)];  // clamp tail
  __syncthreads();

  const float* w1b = W1 + (size_t)n_block * DDIM;
  const float* v1b = V1 + (size_t)n_block * DDIM;

  // Prologue: stage tile k=0 into buffer 0.
  {
    F32Tile tA, tB1, tB2;
    load_f32_tile_gather(tA, X, toks, 0, tid);
    load_f32_tile(tB1, w1b, DDIM, tid);
    load_f32_tile(tB2, v1b, DDIM, tid);
    store_tile_bf16(sA[0],  tA,  tid);
    store_tile_bf16(sB1[0], tB1, tid);
    store_tile_bf16(sB2[0], tB2, tid);
  }
  __syncthreads();

  v8f z = {};
  v8f acc1[4][2], acc2[4][2];
#pragma unroll
  for (int mi = 0; mi < 4; ++mi)
#pragma unroll
    for (int n = 0; n < 2; ++n) { acc1[mi][n] = z; acc2[mi][n] = z; }

  int kb = 0;
  for (int k0 = 0; k0 < DDIM; k0 += BK, kb ^= 1) {
    const bool more = (k0 + BK) < DDIM;
    F32Tile nA, nB1, nB2;
    if (more) {                                        // prefetch tile k+1
      load_f32_tile_gather(nA, X, toks, k0 + BK, tid);
      load_f32_tile(nB1, w1b + (k0 + BK), DDIM, tid);
      load_f32_tile(nB2, v1b + (k0 + BK), DDIM, tid);
    }

    const __bf16* cA  = sA[kb];
    const __bf16* cB1 = sB1[kb];
    const __bf16* cB2 = sB2[kb];
    v16bf a[4];
#pragma unroll
    for (int mi = 0; mi < 4; ++mi)
      a[mi] = frag_a_lds(cA, waveM * 64 + mi * 16, lane);
#pragma unroll
    for (int n = 0; n < 2; ++n) {
      v16bf b1 = frag_b_lds(cB1, waveN * 32 + n * 16, lane);
      v16bf b2 = frag_b_lds(cB2, waveN * 32 + n * 16, lane);
#pragma unroll
      for (int mi = 0; mi < 4; ++mi) {
        acc1[mi][n] = __builtin_amdgcn_wmma_f32_16x16x32_bf16(
            false, a[mi], false, b1, (short)0, acc1[mi][n], false, false);
        acc2[mi][n] = __builtin_amdgcn_wmma_f32_16x16x32_bf16(
            false, a[mi], false, b2, (short)0, acc2[mi][n], false, false);
      }
    }

    if (more) {                                        // convert + fill other buf
      store_tile_bf16(sA[kb ^ 1],  nA,  tid);
      store_tile_bf16(sB1[kb ^ 1], nB1, tid);
      store_tile_bf16(sB2[kb ^ 1], nB2, tid);
    }
    __syncthreads();   // next buf ready AND current buf fully consumed
  }

  const int lr = lane & 15, hi = lane >> 4;
#pragma unroll
  for (int mi = 0; mi < 4; ++mi)
#pragma unroll
    for (int n = 0; n < 2; ++n)
#pragma unroll
      for (int g = 0; g < 8; ++g) {
        int slot = m_block + waveM * 64 + mi * 16 + g + hi * 8;
        int col  = n_block + waveN * 32 + n * 16 + lr;
        float sv = acc1[mi][n][g];
        float vv = acc2[mi][n][g];
        float hval = (sv / (1.0f + expf(-sv))) * vv;   // silu(s) * v
        Hc[(size_t)slot * IDIM + col] = (__bf16)hval;  // padded slots: finite
      }
}

// ---------------------------------------------------------------------------
// Pass 2 (per expert): out[tok,:] += gate * (Hc[slot,:] @ W2^T), masked scatter.
// ---------------------------------------------------------------------------
__global__ __launch_bounds__(256) void moe_out_kernel(
    const __bf16* __restrict__ Hc, const float* __restrict__ W2,
    const float* __restrict__ gates, const int* __restrict__ cntp,
    const int* __restrict__ list_e, int e, float* __restrict__ out) {
  __shared__ __attribute__((aligned(16))) __bf16 sA[2][TILE_ELEMS];
  __shared__ __attribute__((aligned(16))) __bf16 sB[2][TILE_ELEMS];
  __shared__ int toks[BM];

  const int tid = threadIdx.x, lane = tid & 31, wave = tid >> 5;
  const int waveM = wave & 1, waveN = wave >> 1;
  const int cnt = *cntp;
  const int m_block = blockIdx.y * BM;
  if (m_block >= cnt) return;
  const int n_block = blockIdx.x * BN;                 // over DDIM

  if (tid < BM) {
    int s = m_block + tid;
    toks[tid] = (s < cnt) ? list_e[s] : -1;            // -1 masks padded slots
  }
  __syncthreads();

  const __bf16* hb  = Hc + (size_t)m_block * IDIM;
  const float*  w2b = W2 + (size_t)n_block * IDIM;

  {
    Bf16Tile tA;
    F32Tile  tB;
    load_bf16_tile(tA, hb, IDIM, tid);
    load_f32_tile(tB, w2b, IDIM, tid);
    store_tile_copy(sA[0], tA, tid);
    store_tile_bf16(sB[0], tB, tid);
  }
  __syncthreads();

  v8f z = {};
  v8f acc[4][2];
#pragma unroll
  for (int mi = 0; mi < 4; ++mi)
#pragma unroll
    for (int n = 0; n < 2; ++n) acc[mi][n] = z;

  int kb = 0;
  for (int k0 = 0; k0 < IDIM; k0 += BK, kb ^= 1) {
    const bool more = (k0 + BK) < IDIM;
    Bf16Tile nA;
    F32Tile  nB;
    if (more) {
      load_bf16_tile(nA, hb + (k0 + BK), IDIM, tid);
      load_f32_tile(nB, w2b + (k0 + BK), IDIM, tid);
    }

    const __bf16* cA = sA[kb];
    const __bf16* cB = sB[kb];
    v16bf a[4];
#pragma unroll
    for (int mi = 0; mi < 4; ++mi)
      a[mi] = frag_a_lds(cA, waveM * 64 + mi * 16, lane);
#pragma unroll
    for (int n = 0; n < 2; ++n) {
      v16bf b = frag_b_lds(cB, waveN * 32 + n * 16, lane);
#pragma unroll
      for (int mi = 0; mi < 4; ++mi)
        acc[mi][n] = __builtin_amdgcn_wmma_f32_16x16x32_bf16(
            false, a[mi], false, b, (short)0, acc[mi][n], false, false);
    }

    if (more) {
      store_tile_copy(sA[kb ^ 1], nA, tid);
      store_tile_bf16(sB[kb ^ 1], nB, tid);
    }
    __syncthreads();
  }

  const int lr = lane & 15, hi = lane >> 4;
#pragma unroll
  for (int mi = 0; mi < 4; ++mi)
#pragma unroll
    for (int n = 0; n < 2; ++n)
#pragma unroll
      for (int g = 0; g < 8; ++g) {
        int lrow = waveM * 64 + mi * 16 + g + hi * 8;  // slot within block
        int tok  = toks[lrow];
        if (tok >= 0) {
          int col = n_block + waveN * 32 + n * 16 + lr;
          float gate = gates[(size_t)tok * NEXP + e];
          out[(size_t)tok * DDIM + col] += gate * acc[mi][n][g];
        }
      }
}

// ---------------------------------------------------------------------------
// Launcher. Experts sequential so each expert's ~100MB weight set stays hot
// in the 192MB L2. Workspace: [counts | gates | lists | compact H (bf16)].
// ---------------------------------------------------------------------------
extern "C" void kernel_launch(void* const* d_in, const int* in_sizes, int n_in,
                              void* d_out, int out_size, void* d_ws, size_t ws_size,
                              hipStream_t stream) {
  const float* x  = (const float*)d_in[0];
  const float* wr = (const float*)d_in[1];
  const float* w1 = (const float*)d_in[2];
  const float* v1 = (const float*)d_in[3];
  const float* w2 = (const float*)d_in[4];
  float* out = (float*)d_out;

  int*    counts = (int*)d_ws;                                      // 32 B (pad 256)
  float*  gates  = (float*)((char*)d_ws + 256);                     // 128 KB
  int*    list   = (int*)((char*)d_ws + 256 + 4 * T_TOK * NEXP);    // 128 KB
  __bf16* Hc     = (__bf16*)((char*)d_ws + 256 + 8 * T_TOK * NEXP); // 32 MB

  moe_zero_kernel<<<(T_TOK * DDIM / 4) / 256, 256, 0, stream>>>(out, counts);
  moe_router_kernel<<<T_TOK / 8, 256, 0, stream>>>(x, wr, gates, counts, list);

  for (int e = 0; e < NEXP; ++e) {
    moe_h_kernel<<<dim3(IDIM / BN, T_TOK / BM), 256, 0, stream>>>(
        x, w1 + (size_t)e * IDIM * DDIM, v1 + (size_t)e * IDIM * DDIM,
        counts + e, list + (size_t)e * T_TOK, Hc);
    moe_out_kernel<<<dim3(DDIM / BN, T_TOK / BM), 256, 0, stream>>>(
        Hc, w2 + (size_t)e * DDIM * IDIM, gates,
        counts + e, list + (size_t)e * T_TOK, e, out);
  }
}
